// GCN_45200235823214
// MI455X (gfx1250) — matrix operand order repaired and checked by reading, not compiled, not run on hardware
//
#include <hip/hip_runtime.h>
#include <hip/hip_bf16.h>

#define N_NODE  100000
#define N_EDGE  3200000
#define E_TOT   (N_EDGE + N_NODE)   // edges + self loops
#define HEADS   8
#define OUTC    8
#define HC      64
#define BATCH   16384
#define NEG_SLOPE 0.2f

typedef __attribute__((ext_vector_type(16))) __bf16 v16bf;
typedef __attribute__((ext_vector_type(8)))  float  v8f;

// ---------- helpers ----------
__device__ __forceinline__ unsigned fenc(float x) {
    unsigned b = __float_as_uint(x);
    return (b & 0x80000000u) ? ~b : (b | 0x80000000u);   // monotonic f32 -> u32
}
__device__ __forceinline__ float fdec(unsigned u) {
    unsigned b = (u & 0x80000000u) ? (u & 0x7FFFFFFFu) : ~u;
    return __uint_as_float(b);
}
__device__ __forceinline__ float lrelu(float x) { return x > 0.f ? x : NEG_SLOPE * x; }

// ---------- zero init ----------
__global__ void zero_kernel(unsigned* p, int n) {
    int t = blockIdx.x * blockDim.x + threadIdx.x;
    if (t < n) p[t] = 0u;
}

// ---------- 1) h = emb @ lin_w  via bf16 WMMA (K=32 in ONE wmma) ----------
__global__ void __launch_bounds__(32) lin_wmma_kernel(const float* __restrict__ emb,
                                                      const float* __restrict__ lin_w,
                                                      float* __restrict__ h) {
    const int lane = threadIdx.x;
    const int half = lane >> 4;
    const int col  = lane & 15;           // A: row M ; B/C/D: col N
    const int m0   = blockIdx.x * 16;     // 100000/16 = 6250 exact

    // A fragment: 16x32 bf16, M = lane&15, K per ISA layout
    v16bf a;
#pragma unroll
    for (int e = 0; e < 16; ++e) {
        int k = ((e >> 3) << 4) + (half << 3) + (e & 7);
        a[e] = (__bf16)emb[(m0 + col) * 32 + k];
    }
#pragma unroll
    for (int nt = 0; nt < 4; ++nt) {      // HC=64 -> 4 N-tiles
        int n0 = nt * 16;
        v16bf b;
#pragma unroll
        for (int e = 0; e < 16; ++e) {
            int k = ((e >> 3) << 4) + (half << 3) + (e & 7);      // B: K per ISA layout (same as A), N = lane&15
            b[e] = (__bf16)lin_w[k * HC + n0 + col];
        }
        v8f c = {};
        c = __builtin_amdgcn_wmma_f32_16x16x32_bf16(false, a, false, b,
                                                    (short)0, c, false, false);
#pragma unroll
        for (int r = 0; r < 8; ++r) {
            int m = m0 + (half << 3) + r;
            h[m * HC + n0 + col] = c[r];
        }
    }
}

// ---------- 2) per-(node,head) attention scores ----------
__global__ void att_kernel(const float* __restrict__ h,
                           const float* __restrict__ att_src,
                           const float* __restrict__ att_dst,
                           float* __restrict__ a_src, float* __restrict__ a_dst) {
    int t = blockIdx.x * blockDim.x + threadIdx.x;
    if (t >= N_NODE * HEADS) return;
    int n = t >> 3, hd = t & 7;
    const float* hp = h + n * HC + hd * OUTC;
    const float* ws = att_src + hd * OUTC;
    const float* wd = att_dst + hd * OUTC;
    float s = 0.f, d = 0.f;
#pragma unroll
    for (int c = 0; c < OUTC; ++c) { float v = hp[c]; s += v * ws[c]; d += v * wd[c]; }
    a_src[t] = s;
    a_dst[t] = d;
}

// ---------- 3) segment max over incoming edges (encoded u32 atomicMax) ----------
__global__ void edge_max_kernel(const int* __restrict__ ei,
                                const float* __restrict__ a_src,
                                const float* __restrict__ a_dst,
                                unsigned* __restrict__ amax) {
    int t = blockIdx.x * blockDim.x + threadIdx.x;
    if (t >= E_TOT) return;
    int s, d;
    if (t < N_EDGE) { s = ei[t]; d = ei[N_EDGE + t]; }
    else            { s = t - N_EDGE; d = s; }          // self loop
#pragma unroll
    for (int hd = 0; hd < HEADS; ++hd) {
        float al = lrelu(a_src[s * HEADS + hd] + a_dst[d * HEADS + hd]);
        atomicMax(&amax[d * HEADS + hd], fenc(al));
    }
}

// ---------- 4) exp / denom / weighted message scatter-add ----------
__global__ void edge_acc_kernel(const int* __restrict__ ei,
                                const float* __restrict__ a_src,
                                const float* __restrict__ a_dst,
                                const unsigned* __restrict__ amax,
                                const float* __restrict__ h,
                                float* __restrict__ denom,
                                float* __restrict__ acc) {
    int t = blockIdx.x * blockDim.x + threadIdx.x;   // (edge, head) pairs
    if (t >= E_TOT * HEADS) return;
    int e = t >> 3, hd = t & 7;
    int s, d;
    if (e < N_EDGE) { s = ei[e]; d = ei[N_EDGE + e]; }
    else            { s = e - N_EDGE; d = s; }
    float al = lrelu(a_src[s * HEADS + hd] + a_dst[d * HEADS + hd]);
    float ex = __expf(al - fdec(amax[d * HEADS + hd]));
    atomicAdd(&denom[d * HEADS + hd], ex);
    const float* hp = h + s * HC + hd * OUTC;
    float* ap = acc + d * HC + hd * OUTC;
#pragma unroll
    for (int c = 0; c < OUTC; ++c) atomicAdd(&ap[c], ex * hp[c]);
}

// ---------- 5) normalize + bias + relu ----------
__global__ void node_final_kernel(const float* __restrict__ acc,
                                  const float* __restrict__ denom,
                                  const float* __restrict__ conv_b,
                                  float* __restrict__ hrelu) {
    int t = blockIdx.x * blockDim.x + threadIdx.x;
    if (t >= N_NODE * HC) return;
    int n = t >> 6, j = t & 63, hd = j >> 3;
    float v = acc[t] / (denom[n * HEADS + hd] + 1e-16f) + conv_b[j];
    hrelu[t] = v > 0.f ? v : 0.f;
}

// ---------- 6) gather + 4-layer MLP, all GEMM layers in bf16 WMMA ----------
__global__ void __launch_bounds__(32) mlp_wmma_kernel(
        const int* __restrict__ users, const int* __restrict__ items,
        const float* __restrict__ hrelu,
        const float* __restrict__ w1, const float* __restrict__ b1,
        const float* __restrict__ w2, const float* __restrict__ b2,
        const float* __restrict__ w3, const float* __restrict__ b3,
        const float* __restrict__ w4, const float* __restrict__ b4,
        float* __restrict__ out) {
    __shared__ float z [16 * 128];
    __shared__ float z1[16 * 64];
    __shared__ float z2[16 * 32];
    __shared__ float z3[16 * 16];
    const int lane = threadIdx.x;
    const int half = lane >> 4;
    const int col  = lane & 15;
    const int b0   = blockIdx.x * 16;     // 16384/16 = 1024 blocks

    // gather z = [hrelu[user] | hrelu[item]]  (16 rows x 128)
    for (int idx = lane; idx < 16 * 128; idx += 32) {
        int r = idx >> 7, c = idx & 127;
        int node = (c < 64) ? users[b0 + r] : items[b0 + r];
        z[idx] = hrelu[node * HC + (c & 63)];
    }
    __syncthreads();

    // ---- layer1: [16x128] @ [128x64], K-steps = 4 ----
    v16bf a1[4];
#pragma unroll
    for (int ks = 0; ks < 4; ++ks)
#pragma unroll
        for (int e = 0; e < 16; ++e) {
            int k = ks * 32 + ((e >> 3) << 4) + (half << 3) + (e & 7);
            a1[ks][e] = (__bf16)z[col * 128 + k];
        }
#pragma unroll
    for (int nt = 0; nt < 4; ++nt) {
        int n0 = nt * 16;
        v8f c = {};
#pragma unroll
        for (int ks = 0; ks < 4; ++ks) {
            v16bf bf;
#pragma unroll
            for (int e = 0; e < 16; ++e) {
                int k = ks * 32 + ((e >> 3) << 4) + (half << 3) + (e & 7);
                bf[e] = (__bf16)w1[k * 64 + n0 + col];
            }
            c = __builtin_amdgcn_wmma_f32_16x16x32_bf16(false, a1[ks], false, bf,
                                                        (short)0, c, false, false);
        }
#pragma unroll
        for (int r = 0; r < 8; ++r) {
            int m = (half << 3) + r;
            float v = c[r] + b1[n0 + col];
            z1[m * 64 + n0 + col] = v > 0.f ? v : 0.f;
        }
    }
    __syncthreads();

    // ---- layer2: [16x64] @ [64x32], K-steps = 2 ----
    v16bf a2[2];
#pragma unroll
    for (int ks = 0; ks < 2; ++ks)
#pragma unroll
        for (int e = 0; e < 16; ++e) {
            int k = ks * 32 + ((e >> 3) << 4) + (half << 3) + (e & 7);
            a2[ks][e] = (__bf16)z1[col * 64 + k];
        }
#pragma unroll
    for (int nt = 0; nt < 2; ++nt) {
        int n0 = nt * 16;
        v8f c = {};
#pragma unroll
        for (int ks = 0; ks < 2; ++ks) {
            v16bf bf;
#pragma unroll
            for (int e = 0; e < 16; ++e) {
                int k = ks * 32 + ((e >> 3) << 4) + (half << 3) + (e & 7);
                bf[e] = (__bf16)w2[k * 32 + n0 + col];
            }
            c = __builtin_amdgcn_wmma_f32_16x16x32_bf16(false, a2[ks], false, bf,
                                                        (short)0, c, false, false);
        }
#pragma unroll
        for (int r = 0; r < 8; ++r) {
            int m = (half << 3) + r;
            float v = c[r] + b2[n0 + col];
            z2[m * 32 + n0 + col] = v > 0.f ? v : 0.f;
        }
    }
    __syncthreads();

    // ---- layer3: [16x32] @ [32x16], single WMMA ----
    {
        v16bf a3, bf;
#pragma unroll
        for (int e = 0; e < 16; ++e) {
            int k = ((e >> 3) << 4) + (half << 3) + (e & 7);
            a3[e] = (__bf16)z2[col * 32 + k];
        }
#pragma unroll
        for (int e = 0; e < 16; ++e) {
            int k = ((e >> 3) << 4) + (half << 3) + (e & 7);
            bf[e] = (__bf16)w3[k * 16 + col];
        }
        v8f c = {};
        c = __builtin_amdgcn_wmma_f32_16x16x32_bf16(false, a3, false, bf,
                                                    (short)0, c, false, false);
#pragma unroll
        for (int r = 0; r < 8; ++r) {
            int m = (half << 3) + r;
            float v = c[r] + b3[col];
            z3[m * 16 + col] = v > 0.f ? v : 0.f;
        }
    }
    __syncthreads();

    // ---- layer4: [16x16] @ [16x1] scalar ----
    if (lane < 16) {
        float s = b4[0];
#pragma unroll
        for (int k = 0; k < 16; ++k) s += z3[lane * 16 + k] * w4[k];
        out[b0 + lane] = s;
    }
}

// ---------- launcher ----------
extern "C" void kernel_launch(void* const* d_in, const int* in_sizes, int n_in,
                              void* d_out, int out_size, void* d_ws, size_t ws_size,
                              hipStream_t stream) {
    const int*   users = (const int*)d_in[0];
    const int*   items = (const int*)d_in[1];
    const int*   ei    = (const int*)d_in[2];
    const float* emb   = (const float*)d_in[3];
    const float* lin_w = (const float*)d_in[4];
    const float* att_s = (const float*)d_in[5];
    const float* att_d = (const float*)d_in[6];
    const float* convb = (const float*)d_in[7];
    const float* w1 = (const float*)d_in[8],  *b1 = (const float*)d_in[9];
    const float* w2 = (const float*)d_in[10], *b2 = (const float*)d_in[11];
    const float* w3 = (const float*)d_in[12], *b3 = (const float*)d_in[13];
    const float* w4 = (const float*)d_in[14], *b4 = (const float*)d_in[15];

    // workspace layout (floats). hrelu aliases h (h is dead after edge_acc).
    float*    f_h     = (float*)d_ws;                       // N*64  (later: hrelu)
    float*    f_asrc  = f_h    + (size_t)N_NODE * HC;       // N*8
    float*    f_adst  = f_asrc + (size_t)N_NODE * HEADS;    // N*8
    unsigned* u_amax  = (unsigned*)(f_adst + (size_t)N_NODE * HEADS);  // N*8
    float*    f_denom = (float*)(u_amax + (size_t)N_NODE * HEADS);     // N*8
    float*    f_acc   = f_denom + (size_t)N_NODE * HEADS;   // N*64

    // zero amax+denom+acc (contiguous): u32 0 == f32 0.0 == amax identity
    int nz = N_NODE * HEADS * 2 + N_NODE * HC;
    zero_kernel<<<(nz + 255) / 256, 256, 0, stream>>>(u_amax, nz);

    lin_wmma_kernel<<<N_NODE / 16, 32, 0, stream>>>(emb, lin_w, f_h);

    att_kernel<<<(N_NODE * HEADS + 255) / 256, 256, 0, stream>>>(
        f_h, att_s, att_d, f_asrc, f_adst);

    edge_max_kernel<<<(E_TOT + 255) / 256, 256, 0, stream>>>(
        ei, f_asrc, f_adst, u_amax);

    edge_acc_kernel<<<(E_TOT * HEADS + 255) / 256, 256, 0, stream>>>(
        ei, f_asrc, f_adst, u_amax, f_h, f_denom, f_acc);

    node_final_kernel<<<(N_NODE * HC + 255) / 256, 256, 0, stream>>>(
        f_acc, f_denom, convb, f_h);   // writes hrelu over h

    mlp_wmma_kernel<<<BATCH / 16, 32, 0, stream>>>(
        users, items, f_h, w1, b1, w2, b2, w3, b3, w4, b4, (float*)d_out);
}
